// NonLocal2D_s_wo_activation_20882130993409
// MI455X (gfx1250) — matrix operand order repaired and checked by reading, not compiled, not run on hardware
//
#include <hip/hip_runtime.h>

// ---------------------------------------------------------------- constants
#define BATCH 4
#define CCH   256   // in channels
#define ICH   128   // inter channels
#define NPOS  4096  // h*w = 64*64
#define EPSBN 1e-5f
#define KB    64    // attention key-block staged in LDS

typedef __attribute__((ext_vector_type(16))) __bf16 v16bf;
typedef __attribute__((ext_vector_type(8)))  __bf16 v8bf;
typedef __attribute__((ext_vector_type(8)))  float  v8f;
typedef __attribute__((ext_vector_type(4)))  int    v4i;

union V16U { v16bf v; v8bf h[2]; };

static __device__ inline v8f wmma_bf16(v16bf a, v16bf b, v8f c) {
  return __builtin_amdgcn_wmma_f32_16x16x32_bf16(false, a, false, b,
                                                 (short)0, c, false, false);
}

// ds_swizzle XOR butterfly within 32 lanes (and_mask=0x1f, or=0, xor=MASK)
template <int MASK>
static __device__ inline float swz(float x) {
  return __int_as_float(
      __builtin_amdgcn_ds_swizzle(__float_as_int(x), (MASK << 10) | 0x1F));
}

static __device__ inline v8f zero8() {
  v8f z = {0.f, 0.f, 0.f, 0.f, 0.f, 0.f, 0.f, 0.f};
  return z;
}

// 16-byte async global->LDS copy (ASYNCcnt tracked)
static __device__ inline void async_cp16(const __bf16* gp, __bf16* lp) {
#if __has_builtin(__builtin_amdgcn_global_load_async_to_lds_b128)
  __builtin_amdgcn_global_load_async_to_lds_b128(
      (__attribute__((address_space(1))) v4i*)gp,
      (__attribute__((address_space(3))) v4i*)lp, 0, 0);
#else
  asm volatile("global_load_async_to_lds_b128 %0, %1, off"
               :
               : "v"((unsigned)(unsigned long)(
                     (__attribute__((address_space(3))) char*)lp)),
                 "v"((const __attribute__((address_space(1))) void*)gp)
               : "memory");
#endif
}

static __device__ inline void wait_async0() {
#if __has_builtin(__builtin_amdgcn_s_wait_asynccnt)
  __builtin_amdgcn_s_wait_asynccnt(0);
#else
  asm volatile("s_wait_asynccnt 0x0" ::: "memory");
#endif
}

// ------------------------------------------------------------ workspace map
#define OFF_XT   ((size_t)0)                       // (B,N,C) bf16  8 MB
#define OFF_WQ   ((size_t)8388608)                 // theta_w bf16 (I,C)
#define OFF_WK   ((size_t)8454144)                 // phi_w
#define OFF_WG   ((size_t)8519680)                 // g_w
#define OFF_WO   ((size_t)8585216)                 // out_w bf16 (C,I)
#define OFF_INV  ((size_t)8650752)                 // f32 [C]
#define OFF_B2   ((size_t)8654848)                 // f32 [C]
#define OFF_Q    ((size_t)8658944)                 // (B,N,I) bf16 4 MB
#define OFF_K    ((size_t)12853248)                // (B,N,I) bf16
#define OFF_V    ((size_t)17047552)                // (B,I,N) bf16
#define OFF_Y    ((size_t)21241856)                // (B,N,I) bf16

// ------------------------------------------------------- kernel 1: prep/BN
__global__ __launch_bounds__(256) void prep_kernel(
    const float* gw, const float* tw, const float* pw, const float* ow,
    const float* gamma, const float* beta, const float* mean, const float* var,
    __bf16* Wq, __bf16* Wk, __bf16* Wg, __bf16* Wo, float* inv, float* b2) {
  const int WSZ = ICH * CCH;
  int t = blockIdx.x * blockDim.x + threadIdx.x;
  if (t < WSZ)               Wq[t]           = (__bf16)tw[t];
  else if (t < 2 * WSZ)      Wk[t - WSZ]     = (__bf16)pw[t - WSZ];
  else if (t < 3 * WSZ)      Wg[t - 2 * WSZ] = (__bf16)gw[t - 2 * WSZ];
  else if (t < 4 * WSZ)      Wo[t - 3 * WSZ] = (__bf16)ow[t - 3 * WSZ];
  if (t < CCH) {
    float iv = gamma[t] * rsqrtf(var[t] + EPSBN);
    inv[t] = iv;
    b2[t]  = beta[t] - mean[t] * iv;
  }
}

// -------------------------------------------- kernel 2: x (B,C,N) -> Xt bf16
__global__ __launch_bounds__(256) void xpose_kernel(const float* x, __bf16* Xt) {
  __shared__ float tile[32][33];
  int b = blockIdx.z, cb = blockIdx.y * 32, pb = blockIdx.x * 32;
  int tx = threadIdx.x, ty = threadIdx.y;  // 32 x 8
#pragma unroll
  for (int j = 0; j < 4; ++j)
    tile[ty + 8 * j][tx] =
        x[((size_t)b * CCH + cb + ty + 8 * j) * NPOS + pb + tx];
  __syncthreads();
#pragma unroll
  for (int j = 0; j < 4; ++j)
    Xt[((size_t)b * NPOS + pb + ty + 8 * j) * CCH + cb + tx] =
        (__bf16)tile[tx][ty + 8 * j];
}

// ------------------------- kernel 3: projections Q/K (N,I) + V (I,N), bf16
__global__ __launch_bounds__(128) void proj_kernel(
    const __bf16* Xt, const __bf16* Wq, const __bf16* Wk, const __bf16* Wg,
    const float* tb, const float* pb, const float* gb,
    __bf16* Q, __bf16* K, __bf16* V) {
  int lane = threadIdx.x & 31, w = threadIdx.x >> 5;
  int half = lane >> 4, ln = lane & 15;
  int pbase = (blockIdx.x * 4 + w) * 16;
  int pj = blockIdx.y % 3, b = blockIdx.y / 3;
  const __bf16* W    = (pj == 0) ? Wq : (pj == 1) ? Wk : Wg;
  const float*  bias = (pj == 0) ? tb : (pj == 1) ? pb : gb;

  const __bf16* xrow = Xt + (size_t)b * NPOS * CCH + (size_t)(pbase + ln) * CCH;
  v8f acc[8];
#pragma unroll
  for (int it = 0; it < 8; ++it) acc[it] = zero8();

#pragma unroll
  for (int cb = 0; cb < CCH; cb += 32) {
    V16U ua;
    ua.h[0] = *(const v8bf*)(xrow + cb + 16 * half);
    ua.h[1] = *(const v8bf*)(xrow + cb + 16 * half + 8);
#pragma unroll
    for (int it = 0; it < 8; ++it) {
      const __bf16* wp = W + (size_t)(it * 16 + ln) * CCH + cb + 16 * half;
      acc[it] = wmma_bf16(ua.v, *(const v16bf*)wp, acc[it]);
    }
  }

  if (pj < 2) {  // Q / K : (b, p, i) row-major
    __bf16* dst = ((pj == 0) ? Q : K) + (size_t)b * NPOS * ICH;
#pragma unroll
    for (int it = 0; it < 8; ++it) {
      float bv = bias[it * 16 + ln];
#pragma unroll
      for (int r = 0; r < 8; ++r) {
        int p = pbase + r + 8 * half;
        dst[(size_t)p * ICH + it * 16 + ln] = (__bf16)(acc[it][r] + bv);
      }
    }
  } else {       // V (=g) : (b, i, p)
    __bf16* dst = V + (size_t)b * ICH * NPOS;
#pragma unroll
    for (int it = 0; it < 8; ++it) {
      float bv = bias[it * 16 + ln];
#pragma unroll
      for (int r = 0; r < 8; ++r) {
        int p = pbase + r + 8 * half;
        dst[(size_t)(it * 16 + ln) * NPOS + p] = (__bf16)(acc[it][r] + bv);
      }
    }
  }
}

// -------------------- kernel 4: fused flash attention with LDS staging
// 4 waves/block share one 64-key block of K,V staged via async LDS loads.
__global__ __launch_bounds__(128) void attn_kernel(
    const __bf16* Q, const __bf16* K, const __bf16* V, __bf16* Y) {
  __shared__ __align__(16) __bf16 Klds[KB][136];    // [key][i], padded pitch
  __shared__ __align__(16) __bf16 Vlds[ICH][72];    // [i][key], padded pitch
  __shared__ __align__(16) __bf16 pbuf[4][16][72];  // per-wave P repack tile
  int tid = threadIdx.x;
  int lane = tid & 31, w = tid >> 5;
  int half = lane >> 4, ln = lane & 15;
  int qbase = (blockIdx.x * 4 + w) * 16;
  int b = blockIdx.y;

  // Q fragments for the whole I=128 reduction (4 A-tiles of 16x32)
  const __bf16* qrow = Q + ((size_t)b * NPOS + qbase + ln) * ICH;
  v16bf qa[4];
#pragma unroll
  for (int kc = 0; kc < 4; ++kc) {
    V16U u;
    u.h[0] = *(const v8bf*)(qrow + 32 * kc + 16 * half);
    u.h[1] = *(const v8bf*)(qrow + 32 * kc + 16 * half + 8);
    qa[kc] = u.v;
  }

  v8f o[8];
  float m[8], l[8];
#pragma unroll
  for (int it = 0; it < 8; ++it) o[it] = zero8();
#pragma unroll
  for (int r = 0; r < 8; ++r) { m[r] = -1e30f; l[r] = 0.f; }

  const __bf16* Kb = K + (size_t)b * NPOS * ICH;
  const __bf16* Vb = V + (size_t)b * ICH * NPOS;

  for (int kb = 0; kb < NPOS; kb += KB) {
    __syncthreads();  // previous block's LDS reads are done
    // stage K block: KB x ICH elems = 1024 x 16B chunks
#pragma unroll
    for (int j = 0; j < 8; ++j) {
      int id = tid + 128 * j;
      int row = id >> 4, off = (id & 15) * 8;
      async_cp16(Kb + (size_t)(kb + row) * ICH + off, &Klds[row][off]);
    }
    // stage V block: ICH x KB elems = 1024 x 16B chunks
#pragma unroll
    for (int j = 0; j < 8; ++j) {
      int id = tid + 128 * j;
      int row = id >> 3, off = (id & 7) * 8;
      async_cp16(Vb + (size_t)row * NPOS + kb + off, &Vlds[row][off]);
    }
    wait_async0();
    __syncthreads();

    // S = Q * K^T for 64 keys (4 column tiles)
    v8f s[4];
#pragma unroll
    for (int j = 0; j < 4; ++j) s[j] = zero8();
#pragma unroll
    for (int j = 0; j < 4; ++j)
#pragma unroll
      for (int kc = 0; kc < 4; ++kc)
        s[j] = wmma_bf16(qa[kc],
                         *(const v16bf*)&Klds[j * 16 + ln][kc * 32 + 16 * half],
                         s[j]);

    // online softmax over the 64-key block
    float sc[8];
#pragma unroll
    for (int r = 0; r < 8; ++r) {
      float v = fmaxf(fmaxf(s[0][r], s[1][r]), fmaxf(s[2][r], s[3][r]));
      v = fmaxf(v, swz<1>(v)); v = fmaxf(v, swz<2>(v));
      v = fmaxf(v, swz<4>(v)); v = fmaxf(v, swz<8>(v));
      float mn = fmaxf(m[r], v);
      sc[r] = __expf(m[r] - mn);
      m[r] = mn;
      int row = r + 8 * half;
      float rs = 0.f;
#pragma unroll
      for (int j = 0; j < 4; ++j) {
        float p = __expf(s[j][r] - mn);
        rs += p;
        pbuf[w][row][j * 16 + ln] = (__bf16)p;
      }
      rs += swz<1>(rs); rs += swz<2>(rs);
      rs += swz<4>(rs); rs += swz<8>(rs);
      l[r] = l[r] * sc[r] + rs;
    }
#pragma unroll
    for (int it = 0; it < 8; ++it)
#pragma unroll
      for (int r = 0; r < 8; ++r) o[it][r] *= sc[r];

    asm volatile("s_wait_dscnt 0" ::: "memory");  // intra-wave D->A repack fence

    V16U a0, a1;
    const __bf16* prow = &pbuf[w][ln][0];
    a0.h[0] = *(const v8bf*)(prow + 16 * half);
    a0.h[1] = *(const v8bf*)(prow + 16 * half + 8);
    a1.h[0] = *(const v8bf*)(prow + 32 + 16 * half);
    a1.h[1] = *(const v8bf*)(prow + 32 + 16 * half + 8);
#pragma unroll
    for (int it = 0; it < 8; ++it) {
      o[it] = wmma_bf16(a0.v, *(const v16bf*)&Vlds[it * 16 + ln][16 * half],
                        o[it]);
      o[it] = wmma_bf16(a1.v, *(const v16bf*)&Vlds[it * 16 + ln][32 + 16 * half],
                        o[it]);
    }
  }

  // softmax-first then /sqrt(C)=16: fold into normalization
  __bf16* yb = Y + (size_t)b * NPOS * ICH;
#pragma unroll
  for (int it = 0; it < 8; ++it)
#pragma unroll
    for (int r = 0; r < 8; ++r) {
      int p = qbase + r + 8 * half;
      yb[(size_t)p * ICH + it * 16 + ln] =
          (__bf16)(o[it][r] / (l[r] * 16.0f));
    }
}

// --------------------- kernel 5: out = BN(Wout*Y) + x   (fp32 output)
__global__ __launch_bounds__(128) void out_kernel(
    const __bf16* Wo, const __bf16* Y, const float* inv, const float* b2,
    const float* x, float* out) {
  int lane = threadIdx.x & 31, w = threadIdx.x >> 5;
  int half = lane >> 4, ln = lane & 15;
  int cbase = (blockIdx.y * 4 + w) * 16;
  int pbase = blockIdx.x * 16;
  int b = blockIdx.z;

  v8f acc = zero8();
#pragma unroll
  for (int cb = 0; cb < ICH; cb += 32) {
    const __bf16* wrow = Wo + (size_t)(cbase + ln) * ICH;
    V16U ua;
    ua.h[0] = *(const v8bf*)(wrow + cb + 16 * half);
    ua.h[1] = *(const v8bf*)(wrow + cb + 16 * half + 8);
    const __bf16* yp =
        Y + ((size_t)b * NPOS + pbase + ln) * ICH + cb + 16 * half;
    acc = wmma_bf16(ua.v, *(const v16bf*)yp, acc);
  }

  int p = pbase + ln;
#pragma unroll
  for (int r = 0; r < 8; ++r) {
    int c = cbase + r + 8 * half;
    size_t idx = ((size_t)b * CCH + c) * NPOS + p;
    out[idx] = acc[r] * inv[c] + b2[c] + x[idx];
  }
}

// -------------------------------------------------------------------- launch
extern "C" void kernel_launch(void* const* d_in, const int* in_sizes, int n_in,
                              void* d_out, int out_size, void* d_ws,
                              size_t ws_size, hipStream_t stream) {
  const float* x     = (const float*)d_in[0];
  const float* g_w   = (const float*)d_in[1];
  const float* g_b   = (const float*)d_in[2];
  const float* th_w  = (const float*)d_in[3];
  const float* th_b  = (const float*)d_in[4];
  const float* ph_w  = (const float*)d_in[5];
  const float* ph_b  = (const float*)d_in[6];
  const float* out_w = (const float*)d_in[7];
  const float* gamma = (const float*)d_in[8];
  const float* beta  = (const float*)d_in[9];
  const float* mean  = (const float*)d_in[10];
  const float* var   = (const float*)d_in[11];
  float* out = (float*)d_out;

  char* ws = (char*)d_ws;
  __bf16* Xt  = (__bf16*)(ws + OFF_XT);
  __bf16* Wq  = (__bf16*)(ws + OFF_WQ);
  __bf16* Wk  = (__bf16*)(ws + OFF_WK);
  __bf16* Wg  = (__bf16*)(ws + OFF_WG);
  __bf16* Wo  = (__bf16*)(ws + OFF_WO);
  float*  inv = (float*)(ws + OFF_INV);
  float*  b2  = (float*)(ws + OFF_B2);
  __bf16* Qm  = (__bf16*)(ws + OFF_Q);
  __bf16* Km  = (__bf16*)(ws + OFF_K);
  __bf16* Vm  = (__bf16*)(ws + OFF_V);
  __bf16* Ym  = (__bf16*)(ws + OFF_Y);

  prep_kernel<<<512, 256, 0, stream>>>(g_w, th_w, ph_w, out_w, gamma, beta,
                                       mean, var, Wq, Wk, Wg, Wo, inv, b2);
  xpose_kernel<<<dim3(NPOS / 32, CCH / 32, BATCH), dim3(32, 8), 0, stream>>>(
      x, Xt);
  proj_kernel<<<dim3(NPOS / 64, 3 * BATCH), 128, 0, stream>>>(
      Xt, Wq, Wk, Wg, th_b, ph_b, g_b, Qm, Km, Vm);
  attn_kernel<<<dim3(NPOS / 64, BATCH), 128, 0, stream>>>(Qm, Km, Vm, Ym);
  out_kernel<<<dim3(NPOS / 16, CCH / 64, BATCH), 128, 0, stream>>>(
      Wo, Ym, inv, b2, x, out);
}
